// shGCN_23398981829376
// MI455X (gfx1250) — matrix operand order repaired and checked by reading, not compiled, not run on hardware
//
#include <hip/hip_runtime.h>

// ---------------------------------------------------------------------------
// shGCN pipeline for MI455X (gfx1250, wave32, WMMA 16x16x32 f16->f32)
//
//   encoder: conv1d+fc1  ==  x[N,100] @ Weff[100,64] + ebias      (one GEMM)
//   graph:   segment_sum over edges == per-graph dense A[64,64] @ h[64,64]
//   mix:     out = agg@relW^T + h@rootW^T + rel_b                 (WMMA)
//   BN stats via LDS-reduced partial sums + global f32 atomics
//
// v3: division-free, branch-light LDS staging in the encoder (8 threads/row,
// pad region written as aligned 8B zero stores); everything else keeps the
// v2 structure: pre-swizzled weight fragments (32B contiguous v16h loads),
// ds_load_b128 A-fragments, dual h layouts in LDS for the graph stage.
// ---------------------------------------------------------------------------

typedef __attribute__((ext_vector_type(16))) _Float16 v16h;
typedef __attribute__((ext_vector_type(8)))  _Float16 v8h;
typedef __attribute__((ext_vector_type(8)))  float    v8f;

#define N_NODES   32768
#define T_LEN     100
#define KPAD      128
#define D_LAT     64
#define D_HID     128
#define N_GRAPHS  512
#define NPG       64
#define E_TOTAL   (1 << 20)
#define EPG       2048          // E_TOTAL / N_GRAPHS

// ---- WMMA fragment index helpers (per CDNA5 ISA 7.12.2, wave32) ----------
__device__ __forceinline__ int c_m(int lane, int i) { return ((lane >> 4) << 3) + i; }
__device__ __forceinline__ int c_n(int lane)        { return lane & 15; }

// A fragment (16x32 f16, M x K) from a row-major f16 buffer:
// lane<16 holds k {0..15}, lane>=16 holds {16..31}.
// Two contiguous 8-half (16B) chunks -> ds_load_b128 / global_load_b128.
__device__ __forceinline__ v16h load_a_f16(const _Float16* row, int kofs, int lane) {
    const int khalf = (lane & 16) ? 16 : 0;
    union { v16h v; v8h h[2]; } u;
    u.h[0] = *(const v8h*)(row + kofs + khalf);
    u.h[1] = *(const v8h*)(row + kofs + khalf + 8);
    return u.v;
}

// A fragment from a row-major f32 buffer (adjacency counts): aligned float4
// reads + pack-convert.
__device__ __forceinline__ v16h load_a_f32(const float* row, int kofs, int lane) {
    const int khalf = (lane & 16) ? 16 : 0;
    const float4 f0 = *(const float4*)(row + kofs + khalf);
    const float4 f1 = *(const float4*)(row + kofs + khalf + 4);
    const float4 f2 = *(const float4*)(row + kofs + khalf + 8);
    const float4 f3 = *(const float4*)(row + kofs + khalf + 12);
    v16h a;
    a[0]  = (_Float16)f0.x; a[1]  = (_Float16)f0.y; a[2]  = (_Float16)f0.z; a[3]  = (_Float16)f0.w;
    a[4]  = (_Float16)f1.x; a[5]  = (_Float16)f1.y; a[6]  = (_Float16)f1.z; a[7]  = (_Float16)f1.w;
    a[8]  = (_Float16)f2.x; a[9]  = (_Float16)f2.y; a[10] = (_Float16)f2.z; a[11] = (_Float16)f2.w;
    a[12] = (_Float16)f3.x; a[13] = (_Float16)f3.y; a[14] = (_Float16)f3.z; a[15] = (_Float16)f3.w;
    return a;
}

// B fragment (32x16 f16, K x N) from an n-major f16 buffer: contiguous K run
// of 16 -> one 32B v16h load.
__device__ __forceinline__ v16h load_b_nmajor(const _Float16* buf, int n, int kofs,
                                              int kstride, int lane) {
    const int kbase = kofs + ((lane >> 4) << 4);
    return *(const v16h*)(buf + (n + (lane & 15)) * kstride + kbase);
}

// ---------------------------------------------------------------------------
// K1: fold conv1d(K=25,pad=12)+fc1 into Weff; emit all weight matrices
// pre-swizzled into WMMA B-fragment order; encoder bias; zero BN acc.
//   Weff_frag : [nt:4][kb:4][lane:32][i:16]   (K=128 zero-padded, N=64)
//   relW_frag : [nt:8][kb:2][lane:32][i:16]   (K=64, N=128)
// Fragment element (lane,i): n = nt*16 + (lane&15), k = kb*32 + ((lane>>4)<<4) + i
// ---------------------------------------------------------------------------
__global__ __launch_bounds__(256) void k_prep(
        const float* __restrict__ conv_w,  // [64][25]
        const float* __restrict__ conv_b,  // [64]
        const float* __restrict__ fc1_w,   // [100]
        const float* __restrict__ fc1_b,   // [1]
        const float* __restrict__ rel_w,   // [128][64]
        const float* __restrict__ root_w,  // [128][64]
        _Float16* __restrict__ Weff_frag,
        _Float16* __restrict__ relW_frag,
        _Float16* __restrict__ rootW_frag,
        float* __restrict__ ebias,         // [64]
        float* __restrict__ bn_acc)        // [256]
{
    const int tid = threadIdx.x;
    bn_acc[tid] = 0.f;

    // Weff(k, c) = sum_kk conv_w[c,kk] * fc1_w[k - kk + 12]
    for (int idx = tid; idx < 4 * 4 * 32 * 16; idx += 256) {
        int i    = idx & 15;
        int lane = (idx >> 4) & 31;
        int kb   = (idx >> 9) & 3;
        int nt   = idx >> 11;
        int c    = nt * 16 + (lane & 15);
        int k    = kb * 32 + ((lane >> 4) << 4) + i;
        float s = 0.f;
        if (k < T_LEN) {
            #pragma unroll
            for (int kk = 0; kk < 25; ++kk) {
                int t = k - kk + 12;
                if (t >= 0 && t < T_LEN) s += conv_w[c * 25 + kk] * fc1_w[t];
            }
        }
        Weff_frag[idx] = (_Float16)s;
    }
    for (int idx = tid; idx < 8 * 2 * 32 * 16; idx += 256) {
        int i    = idx & 15;
        int lane = (idx >> 4) & 31;
        int kb   = (idx >> 9) & 1;
        int nt   = idx >> 10;
        int n    = nt * 16 + (lane & 15);
        int k    = kb * 32 + ((lane >> 4) << 4) + i;
        relW_frag[idx]  = (_Float16)rel_w[n * D_LAT + k];
        rootW_frag[idx] = (_Float16)root_w[n * D_LAT + k];
    }
    if (tid < D_LAT) {
        float s = 0.f;
        for (int t = 0; t < T_LEN; ++t) s += fc1_w[t];
        ebias[tid] = conv_b[tid] * s + fc1_b[0];
    }
}

// ---------------------------------------------------------------------------
// K2: h[N,64] = x[N,100] @ Weff + ebias.  Block = 8 waves = 32 rows x 64 cols.
// Division-free staging: 8 threads per row, each covers float4 slots
// q = (tid&7)+8j; q<25 converts data, q>=25 writes aligned zero pad
// (covers k in [100,128) exactly).
// ---------------------------------------------------------------------------
__global__ __launch_bounds__(256) void k_encoder(
        const float* __restrict__ x, const _Float16* __restrict__ Weff_frag,
        const float* __restrict__ ebias, float* __restrict__ h)
{
    __shared__ _Float16 xs[32 * KPAD];     // 8 KB
    const int tid  = threadIdx.x;
    const int lane = tid & 31;
    const int w    = tid >> 5;
    const int row_base = blockIdx.x * 32;

    const float4* x4 = (const float4*)(x + row_base * T_LEN);
    const int r  = tid >> 3;               // 0..31 (row)
    const int qb = tid & 7;
    #pragma unroll
    for (int j = 0; j < 4; ++j) {
        const int q = qb + j * 8;          // 0..31 float4 slots (KPAD/4)
        _Float16* p = &xs[r * KPAD + q * 4];
        if (q < 25) {
            float4 f = x4[r * 25 + q];
            p[0] = (_Float16)f.x; p[1] = (_Float16)f.y;
            p[2] = (_Float16)f.z; p[3] = (_Float16)f.w;
        } else {
            p[0] = (_Float16)0.f; p[1] = (_Float16)0.f;
            p[2] = (_Float16)0.f; p[3] = (_Float16)0.f;
        }
    }
    __syncthreads();

    const int mt = (w >> 2) * 16;          // local row tile
    const int nt = (w & 3);                // col tile
    const _Float16* arow  = &xs[(mt + (lane & 15)) * KPAD];
    const _Float16* bfrag = Weff_frag + ((nt * 4) * 32 + lane) * 16;

    v8f c = {};
    #pragma unroll
    for (int kb = 0; kb < 4; ++kb) {
        v16h a = load_a_f16(arow, kb * 32, lane);
        v16h b = *(const v16h*)(bfrag + kb * 32 * 16);
        c = __builtin_amdgcn_wmma_f32_16x16x32_f16(false, a, false, b,
                                                   (short)0, c, false, false);
    }
    const int n = nt * 16 + c_n(lane);
    const float bias = ebias[n];
    #pragma unroll
    for (int i = 0; i < 8; ++i)
        h[(row_base + mt + c_m(lane, i)) * D_LAT + n] = c[i] + bias;
}

// ---------------------------------------------------------------------------
// K3: one block per graph.  Adjacency counts in LDS, then two WMMA stages.
// ---------------------------------------------------------------------------
__global__ __launch_bounds__(256) void k_graphconv(
        const float* __restrict__ h, const int* __restrict__ edge,
        const _Float16* __restrict__ relW_frag, const _Float16* __restrict__ rootW_frag,
        const float* __restrict__ rel_b,
        float* __restrict__ out, float* __restrict__ bn_acc)
{
    __shared__ float    A[NPG * NPG];       // 16 KB adjacency counts
    __shared__ _Float16 hs [NPG * D_LAT];   // 8 KB node-major  (A frags)
    __shared__ _Float16 hsT[D_LAT * NPG];   // 8 KB chan-major  (B frags)
    __shared__ _Float16 ag [NPG * D_LAT];   // 8 KB aggregated, node-major
    __shared__ float    bnloc[256];

    const int g    = blockIdx.x;
    const int tid  = threadIdx.x;
    const int lane = tid & 31;
    const int w    = tid >> 5;

    const int* esrc = edge;
    const int* edst = edge + E_TOTAL;
    __builtin_prefetch(esrc + g, 0, 1);
    __builtin_prefetch(edst + g, 0, 1);

    for (int i = tid; i < NPG * NPG; i += 256) A[i] = 0.f;
    bnloc[tid] = 0.f;
    // stage h tile in both layouts (vectorized float4 reads)
    const float4* h4 = (const float4*)(h + g * NPG * D_LAT);
    for (int idx = tid; idx < NPG * D_LAT / 4; idx += 256) {
        int node = idx >> 4, q = (idx & 15) * 4;
        float4 f = h4[idx];
        _Float16 a0 = (_Float16)f.x, a1 = (_Float16)f.y;
        _Float16 a2 = (_Float16)f.z, a3 = (_Float16)f.w;
        _Float16* p = &hs[node * D_LAT + q];
        p[0] = a0; p[1] = a1; p[2] = a2; p[3] = a3;
        hsT[(q + 0) * NPG + node] = a0;
        hsT[(q + 1) * NPG + node] = a1;
        hsT[(q + 2) * NPG + node] = a2;
        hsT[(q + 3) * NPG + node] = a3;
    }
    __syncthreads();

    // adjacency build: A[dst_local][src_local] += 1   (e = g + 512*i)
    for (int i = tid; i < EPG; i += 256) {
        int e = g + (i << 9);
        int s = esrc[e] & 63;
        int d = edst[e] & 63;
        atomicAdd(&A[d * NPG + s], 1.f);
    }
    __syncthreads();

    // agg = A @ h_g : 4x4 tiles, 2 per wave, K=64
    #pragma unroll
    for (int tt = 0; tt < 2; ++tt) {
        int t  = w * 2 + tt;
        int mt = (t >> 2) * 16, nt = (t & 3) * 16;
        const float* arow = &A[(mt + (lane & 15)) * NPG];
        v8f c = {};
        #pragma unroll
        for (int kb = 0; kb < 2; ++kb) {
            v16h a = load_a_f32(arow, kb * 32, lane);
            v16h b = load_b_nmajor(hsT, nt, kb * 32, NPG, lane);
            c = __builtin_amdgcn_wmma_f32_16x16x32_f16(false, a, false, b,
                                                       (short)0, c, false, false);
        }
        int n = nt + c_n(lane);
        #pragma unroll
        for (int i = 0; i < 8; ++i)
            ag[(mt + c_m(lane, i)) * D_LAT + n] = (_Float16)c[i];
    }
    __syncthreads();

    // out = ag @ relW + hs @ rootW + rel_b : 4x8 tiles, 4 per wave
    #pragma unroll
    for (int tt = 0; tt < 4; ++tt) {
        int t  = w * 4 + tt;
        int mt = (t >> 3) * 16, nt = (t & 7);
        const _Float16* agrow = &ag[(mt + (lane & 15)) * D_LAT];
        const _Float16* hrow  = &hs[(mt + (lane & 15)) * D_LAT];
        const _Float16* relf  = relW_frag  + ((nt * 2) * 32 + lane) * 16;
        const _Float16* rootf = rootW_frag + ((nt * 2) * 32 + lane) * 16;
        v8f c = {};
        #pragma unroll
        for (int kb = 0; kb < 2; ++kb) {
            v16h a = load_a_f16(agrow, kb * 32, lane);
            v16h b = *(const v16h*)(relf + kb * 32 * 16);
            c = __builtin_amdgcn_wmma_f32_16x16x32_f16(false, a, false, b,
                                                       (short)0, c, false, false);
            a = load_a_f16(hrow, kb * 32, lane);
            b = *(const v16h*)(rootf + kb * 32 * 16);
            c = __builtin_amdgcn_wmma_f32_16x16x32_f16(false, a, false, b,
                                                       (short)0, c, false, false);
        }
        int n = nt * 16 + c_n(lane);
        float bias = rel_b[n];
        float s1 = 0.f, s2 = 0.f;
        #pragma unroll
        for (int i = 0; i < 8; ++i) {
            float v = c[i] + bias;
            out[(g * NPG + mt + c_m(lane, i)) * D_HID + n] = v;
            s1 += v;
            s2 += v * v;
        }
        atomicAdd(&bnloc[n], s1);
        atomicAdd(&bnloc[D_HID + n], s2);
    }
    __syncthreads();
    atomicAdd(&bn_acc[tid], bnloc[tid]);
}

// ---------------------------------------------------------------------------
// K4: fold batch statistics into per-channel scale/shift.
// ---------------------------------------------------------------------------
__global__ __launch_bounds__(128) void k_bnstats(
        const float* __restrict__ bn_acc,
        const float* __restrict__ gamma, const float* __restrict__ beta,
        float* __restrict__ scale, float* __restrict__ shift)
{
    int c = threadIdx.x;
    float mean = bn_acc[c] * (1.f / (float)N_NODES);
    float var  = bn_acc[D_HID + c] * (1.f / (float)N_NODES) - mean * mean;
    float sc   = gamma[c] * rsqrtf(var + 1e-5f);
    scale[c] = sc;
    shift[c] = beta[c] - mean * sc;
}

// ---------------------------------------------------------------------------
// K5: normalize, square, mean-pool over 64 nodes, log(clamp), fc2, sigmoid.
// ---------------------------------------------------------------------------
__global__ __launch_bounds__(128) void k_pool_head(
        const float* __restrict__ out,
        const float* __restrict__ scale, const float* __restrict__ shift,
        const float* __restrict__ fc2_w, const float* __restrict__ fc2_b,
        float* __restrict__ y)
{
    __shared__ float pld[D_HID];
    const int g = blockIdx.x;
    const int c = threadIdx.x;
    const float sc = scale[c], sh = shift[c];
    float acc = 0.f;
    #pragma unroll 4
    for (int nd = 0; nd < NPG; ++nd) {
        float v = out[(g * NPG + nd) * D_HID + c] * sc + sh;
        acc += v * v;
    }
    float pooled = acc * (1.f / (float)NPG);
    pld[c] = logf(fmaxf(pooled, 1e-6f));
    __syncthreads();
    if (c < 3) {
        float a = fc2_b[c];
        for (int k = 0; k < D_HID; ++k) a += pld[k] * fc2_w[c * D_HID + k];
        y[g * 3 + c] = 1.f / (1.f + expf(-a));
    }
}

// ---------------------------------------------------------------------------
extern "C" void kernel_launch(void* const* d_in, const int* in_sizes, int n_in,
                              void* d_out, int out_size, void* d_ws, size_t ws_size,
                              hipStream_t stream) {
    const float* x      = (const float*)d_in[0];
    const int*   edge   = (const int*)  d_in[1];
    // d_in[2] = batch (unused: nodes are block-contiguous per graph)
    const float* conv_w = (const float*)d_in[3];
    const float* conv_b = (const float*)d_in[4];
    const float* fc1_w  = (const float*)d_in[5];
    const float* fc1_b  = (const float*)d_in[6];
    const float* rel_w  = (const float*)d_in[7];
    const float* rel_b  = (const float*)d_in[8];
    const float* root_w = (const float*)d_in[9];
    const float* gamma  = (const float*)d_in[10];
    const float* beta   = (const float*)d_in[11];
    const float* fc2_w  = (const float*)d_in[12];
    const float* fc2_b  = (const float*)d_in[13];
    float* y = (float*)d_out;

    char* ws = (char*)d_ws;
    _Float16* Weff_frag  = (_Float16*)(ws);                 // 16 KB
    _Float16* relW_frag  = (_Float16*)(ws + 16384);         // 16 KB
    _Float16* rootW_frag = (_Float16*)(ws + 32768);         // 16 KB
    float*    ebias      = (float*)(ws + 49152);            // 64*4
    float*    bn_acc     = (float*)(ws + 49408);            // 256*4
    float*    bn_sc      = (float*)(ws + 50432);            // 128*4
    float*    bn_sh      = (float*)(ws + 50944);            // 128*4
    float*    h          = (float*)(ws + 65536);            // 8 MB
    float*    outbuf     = (float*)(ws + 65536 + 8388608);  // 16 MB

    k_prep<<<1, 256, 0, stream>>>(conv_w, conv_b, fc1_w, fc1_b, rel_w, root_w,
                                  Weff_frag, relW_frag, rootW_frag, ebias, bn_acc);
    k_encoder<<<N_NODES / 32, 256, 0, stream>>>(x, Weff_frag, ebias, h);
    k_graphconv<<<N_GRAPHS, 256, 0, stream>>>(h, edge, relW_frag, rootW_frag, rel_b,
                                              outbuf, bn_acc);
    k_bnstats<<<1, 128, 0, stream>>>(bn_acc, gamma, beta, bn_sc, bn_sh);
    k_pool_head<<<N_GRAPHS, 128, 0, stream>>>(outbuf, bn_sc, bn_sh,
                                              fc2_w, fc2_b, y);
}
